// DBOT_Sinkhorn_Loss_45372034515278
// MI455X (gfx1250) — compile-verified, matrix operand order checked by reading, not run on hardware
//
#include <hip/hip_runtime.h>
#include <hip/hip_bf16.h>

// DBOT Sinkhorn loss, N=8192, D=512, 5 iterations.
// Plan:
//   ws:  P0 f16 [N*N] (128MB) | imgh f16 | txth f16 | 8 f32 vectors [N] | loss[2]
//   1) cvt features f32->f16
//   2) WMMA GEMM: P0 = exp(scale*img@txt.T - 1)  (f16 out, f32 accum)
//   3) 5x { dual-pass rowsum/colsum (one read serves BOTH sinkhorn directions),
//           u-update, dual-pass, v-update }  -- scaling-vector Sinkhorn
//   4) online-softmax CE over rows (dir A) and columns (dir B), atomic loss accum
//   5) out = (lossA+lossB)/(2N)

#define NN 8192
#define DD 512

typedef __attribute__((ext_vector_type(16))) _Float16 v16h;
typedef __attribute__((ext_vector_type(8)))  _Float16 v8h;
typedef __attribute__((ext_vector_type(4)))  _Float16 v4h;
typedef __attribute__((ext_vector_type(8)))  float    v8f;
typedef __attribute__((ext_vector_type(4)))  float    v4f;

union FragU { v16h v; v8h h[2]; };

// ---------------- feature convert f32 -> f16 ----------------
__global__ void __launch_bounds__(256)
cvt_kernel(const float* __restrict__ img, const float* __restrict__ txt,
           _Float16* __restrict__ imgh, _Float16* __restrict__ txth){
  const float* src = blockIdx.y ? txt : img;
  _Float16* dst    = blockIdx.y ? txth : imgh;
  size_t i = ((size_t)blockIdx.x*256 + threadIdx.x)*4;
  v4f f = *(const v4f*)(src + i);
  v4h h;
#pragma unroll
  for (int k=0;k<4;k++) h[k] = (_Float16)f[k];
  *(v4h*)(dst + i) = h;
}

// ---------------- WMMA GEMM + exp epilogue ----------------
// Per CDNA5 16-bit A-matrix 16x32 layout: lane<16 holds K 0..7 & 16..23,
// lane>=16 holds K 8..15 & 24..31, row m = lane%16..  (B mirrors with N roles)
__device__ inline v16h lds_frag(const _Float16* base, int stride){
  int lane = threadIdx.x & 31;
  const _Float16* p = base + (lane & 15)*stride + ((lane >> 4) << 3);
  FragU f;
  f.h[0] = *(const v8h*)(p);
  f.h[1] = *(const v8h*)(p + 16);
  return f.v;
}

__global__ void __launch_bounds__(256)
gemm_exp_kernel(const _Float16* __restrict__ A, const _Float16* __restrict__ B,
                const float* __restrict__ scale, _Float16* __restrict__ P){
  __shared__ _Float16 As[128][40];   // 128 rows x 32 K, pad to 40
  __shared__ _Float16 Bs[128][40];
  const int t = threadIdx.x;
  const int wave = t >> 5, lane = t & 31;
  const int r0 = blockIdx.y*128, c0 = blockIdx.x*128;
  const int wr = wave >> 1, wc = wave & 1;  // wave tile: 32 rows x 64 cols
  const float s = scale[0];
  const v8f vz = {0.f,0.f,0.f,0.f,0.f,0.f,0.f,0.f};
  v8f acc[2][4];
#pragma unroll
  for (int i=0;i<2;i++)
#pragma unroll
    for (int j=0;j<4;j++) acc[i][j] = vz;

  const int lrow = t >> 1, lk = (t & 1)*16;
  for (int k0 = 0; k0 < DD; k0 += 32){
    *(v8h*)&As[lrow][lk]   = *(const v8h*)(A + (size_t)(r0+lrow)*DD + k0 + lk);
    *(v8h*)&As[lrow][lk+8] = *(const v8h*)(A + (size_t)(r0+lrow)*DD + k0 + lk + 8);
    *(v8h*)&Bs[lrow][lk]   = *(const v8h*)(B + (size_t)(c0+lrow)*DD + k0 + lk);
    *(v8h*)&Bs[lrow][lk+8] = *(const v8h*)(B + (size_t)(c0+lrow)*DD + k0 + lk + 8);
    __syncthreads();
    v16h af[2], bf[4];
#pragma unroll
    for (int i=0;i<2;i++) af[i] = lds_frag(&As[wr*32 + i*16][0], 40);
#pragma unroll
    for (int j=0;j<4;j++) bf[j] = lds_frag(&Bs[wc*64 + j*16][0], 40);
#pragma unroll
    for (int i=0;i<2;i++)
#pragma unroll
      for (int j=0;j<4;j++)
        acc[i][j] = __builtin_amdgcn_wmma_f32_16x16x32_f16(
            false, af[i], false, bf[j], (short)0, acc[i][j], false, false);
    __syncthreads();
  }
  // C layout: VGPR r -> row r + (lane>=16 ? 8 : 0); col = lane%16
  const int mhi = (lane >> 4) << 3;
  const int nc  = lane & 15;
#pragma unroll
  for (int i=0;i<2;i++)
#pragma unroll
    for (int j=0;j<4;j++){
      const int rb = r0 + wr*32 + i*16 + mhi;
      const int cb = c0 + wc*64 + j*16 + nc;
#pragma unroll
      for (int rr=0;rr<8;rr++){
        float c = acc[i][j][rr];
        P[(size_t)(rb+rr)*NN + cb] = (_Float16)__expf(__builtin_fmaf(s, c, -1.0f));
      }
    }
}

// ---------------- dual-direction reduction pass ----------------
// rowOut[r] += sum_c P[r,c]*colW[c] ; colOut[c] += sum_r P[r,c]*rowW[r]
__global__ void __launch_bounds__(256)
pass_dual_kernel(const _Float16* __restrict__ P, const float* __restrict__ colW,
                 const float* __restrict__ rowW, float* __restrict__ rowOut,
                 float* __restrict__ colOut){
  __shared__ float wcs[128], wrs[128];
  __shared__ float red[128*17];      // padded stride 17 -> conflict-free
  const int t = threadIdx.x;
  const int r0 = blockIdx.y*128, c0 = blockIdx.x*128;
  if (t < 128) wcs[t] = colW[c0+t]; else wrs[t-128] = rowW[r0+t-128];
  __syncthreads();
  const int tx = t & 15, ty = t >> 4;     // 16x16 threads, 8x8 elems each
  const int rb = ty*8, cb = tx*8;
  float rAcc[8], cAcc[8];
#pragma unroll
  for (int i=0;i<8;i++){ rAcc[i]=0.f; cAcc[i]=0.f; }
#pragma unroll
  for (int i=0;i<8;i++){
    v8h d = *(const v8h*)(P + (size_t)(r0+rb+i)*NN + c0 + cb);
    float w = wrs[rb+i];
    float ra = 0.f;
#pragma unroll
    for (int k=0;k<8;k++){
      float pv = (float)d[k];
      ra      = __builtin_fmaf(pv, wcs[cb+k], ra);
      cAcc[k] = __builtin_fmaf(pv, w, cAcc[k]);
    }
    rAcc[i] = ra;
  }
  // reduce row partials across tx
#pragma unroll
  for (int i=0;i<8;i++) red[(rb+i)*17 + tx] = rAcc[i];
  __syncthreads();
  if (t < 128){
    float s = 0.f;
#pragma unroll
    for (int x=0;x<16;x++) s += red[t*17+x];
    atomicAdd(&rowOut[r0+t], s);
  }
  __syncthreads();
  // reduce col partials across ty
#pragma unroll
  for (int k=0;k<8;k++) red[(cb+k)*17 + ty] = cAcc[k];
  __syncthreads();
  if (t < 128){
    float s = 0.f;
#pragma unroll
    for (int y=0;y<16;y++) s += red[t*17+y];
    atomicAdd(&colOut[c0+t], s);
  }
}

// ---------------- tiny vector kernels ----------------
__global__ void init_kernel(float* vA, float* vB, float* sA, float* sB,
                            float* cA, float* cB, float* loss){
  int i = blockIdx.x*256 + threadIdx.x;
  vA[i]=1.f; vB[i]=1.f; sA[i]=0.f; sB[i]=0.f; cA[i]=0.f; cB[i]=0.f;
  if (i < 2) loss[i] = 0.f;
}
__global__ void updU_kernel(const float* sA, const float* sB, float* uA, float* uB,
                            float* cA, float* cB){
  int i = blockIdx.x*256 + threadIdx.x;
  uA[i] = 1.f/sA[i]; uB[i] = 1.f/sB[i];
  cA[i] = 0.f; cB[i] = 0.f;
}
__global__ void updV_kernel(float* vA, float* vB, const float* cA, const float* cB,
                            float* sA, float* sB){
  int i = blockIdx.x*256 + threadIdx.x;
  const float bd = 0.1f*(float)NN, bu = 0.9f*(float)NN;
  float va = vA[i]; float c = cA[i]*va;
  float f1 = fmaxf(bd/c, 1.f); va *= f1; c *= f1;
  vA[i] = va * fminf(bu/c, 1.f);
  float vb = vB[i]; float d = cB[i]*vb;
  f1 = fmaxf(bd/d, 1.f); vb *= f1; d *= f1;
  vB[i] = vb * fminf(bu/d, 1.f);
  sA[i] = 0.f; sB[i] = 0.f;
}

// ---------------- CE over rows of diag(u) P0 diag(v) ----------------
__global__ void __launch_bounds__(256)
ce_rows_kernel(const _Float16* __restrict__ P, const float* __restrict__ u,
               const float* __restrict__ v, const int* __restrict__ labels,
               float* __restrict__ lossAcc){
  __shared__ float vs[NN];
  const int t = threadIdx.x;
  for (int i=t;i<NN;i+=256) vs[i]=v[i];
  __syncthreads();
  const int row = blockIdx.x*64 + (t>>2);
  const int cq  = t & 3;
  const float ur = u[row];
  float m = -3.4e38f, s = 0.f;
  for (int j=0;j<256;j++){
    const int cbase = (cq + 4*j)*8;
    v8h d = *(const v8h*)(P + (size_t)row*NN + cbase);
    float val[8]; float cm = -3.4e38f;
#pragma unroll
    for (int k=0;k<8;k++){ val[k] = ur*(float)d[k]*vs[cbase+k]; cm = fmaxf(cm, val[k]); }
    float mn = fmaxf(m, cm);
    float add = 0.f;
#pragma unroll
    for (int k=0;k<8;k++) add += __expf(val[k]-mn);
    s = __builtin_fmaf(s, __expf(m-mn), add);
    m = mn;
  }
#pragma unroll
  for (int off=1; off<4; off<<=1){
    float mo = __shfl_xor(m, off, 32);
    float so = __shfl_xor(s, off, 32);
    float mn = fmaxf(m, mo);
    s = s*__expf(m-mn) + so*__expf(mo-mn);
    m = mn;
  }
  if (cq == 0){
    int lab = labels[row];
    float vd = ur * (float)P[(size_t)row*NN + lab] * vs[lab];
    atomicAdd(&lossAcc[0], m + logf(s) - vd);
  }
}

// ---------------- CE over rows of diag(u) P0^T diag(v) (i.e. columns of P0) ----
__global__ void __launch_bounds__(256)
ce_cols_kernel(const _Float16* __restrict__ P, const float* __restrict__ u,
               const float* __restrict__ v, const int* __restrict__ labels,
               float* __restrict__ lossAcc){
  __shared__ float vs[NN];
  __shared__ float ms[256], ssh[256];
  const int t = threadIdx.x;
  for (int i=t;i<NN;i+=256) vs[i]=v[i];
  __syncthreads();
  const int col = blockIdx.x*64 + (t & 63);
  const int rq  = t >> 6;
  const float uc = u[col];
  float m = -3.4e38f, s = 0.f;
  for (int j=0;j<256;j++){
    float val[8]; float cm = -3.4e38f;
#pragma unroll
    for (int k=0;k<8;k++){
      int r = rq + 4*k + 32*j;
      float pv = (float)P[(size_t)r*NN + col];
      val[k] = uc*pv*vs[r];
      cm = fmaxf(cm, val[k]);
    }
    float mn = fmaxf(m, cm);
    float add = 0.f;
#pragma unroll
    for (int k=0;k<8;k++) add += __expf(val[k]-mn);
    s = __builtin_fmaf(s, __expf(m-mn), add);
    m = mn;
  }
  ms[t]=m; ssh[t]=s;
  __syncthreads();
  if (t < 64){
    float M=ms[t], S=ssh[t];
#pragma unroll
    for (int q=1;q<4;q++){
      float mo=ms[t+64*q], so=ssh[t+64*q];
      float mn=fmaxf(M,mo);
      S = S*__expf(M-mn) + so*__expf(mo-mn);
      M = mn;
    }
    int c2 = blockIdx.x*64 + t;
    int lab = labels[c2];
    float vd = u[c2] * (float)P[(size_t)lab*NN + c2] * vs[lab];
    atomicAdd(&lossAcc[1], M + logf(S) - vd);
  }
}

__global__ void fin_kernel(const float* __restrict__ lossAcc, float* __restrict__ out){
  if (threadIdx.x == 0)
    out[0] = (lossAcc[0] + lossAcc[1]) * (0.5f/(float)NN);
}

extern "C" void kernel_launch(void* const* d_in, const int* in_sizes, int n_in,
                              void* d_out, int out_size, void* d_ws, size_t ws_size,
                              hipStream_t stream) {
  const float* img   = (const float*)d_in[0];
  const float* txt   = (const float*)d_in[1];
  const float* scale = (const float*)d_in[2];
  const int*   labels= (const int*)d_in[3];
  float* out = (float*)d_out;

  char* w = (char*)d_ws;
  _Float16* P0 = (_Float16*)w;                                  // 128 MB
  _Float16* Ah = (_Float16*)(w + (size_t)NN*NN*2);              // 8 MB
  _Float16* Bh = Ah + (size_t)NN*DD;                            // 8 MB
  float* vecs  = (float*)(w + (size_t)NN*NN*2 + (size_t)2*NN*DD*2);
  float *vA=vecs,      *vB=vecs+NN,   *uA=vecs+2*NN, *uB=vecs+3*NN,
        *sA=vecs+4*NN, *sB=vecs+5*NN, *cA=vecs+6*NN, *cB=vecs+7*NN,
        *lossAcc=vecs+8*NN;

  cvt_kernel<<<dim3((NN*DD)/(256*4), 2), 256, 0, stream>>>(img, txt, Ah, Bh);
  gemm_exp_kernel<<<dim3(NN/128, NN/128), 256, 0, stream>>>(Ah, Bh, scale, P0);
  init_kernel<<<NN/256, 256, 0, stream>>>(vA, vB, sA, sB, cA, cB, lossAcc);

  for (int it = 0; it < 5; ++it) {
    // rowsum_A (P0·vA) and rowsum_B (= col-accum of P0 weighted by vB)
    pass_dual_kernel<<<dim3(64,64), 256, 0, stream>>>(P0, vA, vB, sA, sB);
    updU_kernel<<<NN/256, 256, 0, stream>>>(sA, sB, uA, uB, cA, cB);
    // raw colsum_A (uA@P0 -> colOut=cA) and raw colsum_B (rowOut=cB, colW=uB)
    pass_dual_kernel<<<dim3(64,64), 256, 0, stream>>>(P0, uB, uA, cB, cA);
    updV_kernel<<<NN/256, 256, 0, stream>>>(vA, vB, cA, cB, sA, sB);
  }

  ce_rows_kernel<<<NN/64, 256, 0, stream>>>(P0, uA, vA, labels, lossAcc);
  ce_cols_kernel<<<NN/64, 256, 0, stream>>>(P0, uB, vB, labels, lossAcc);
  fin_kernel<<<1, 32, 0, stream>>>(lossAcc, out);
}
